// AttentionHead_72490458022486
// MI455X (gfx1250) — compile-verified, matrix operand order checked by reading, not run on hardware
//
#include <hip/hip_runtime.h>

// Flash attention head for MI455X (gfx1250, wave32, WMMA bf16).
// Kernel 0: convert Wq/Wk/Wv fp32 -> bf16 transposed Wt[z][h][k] (one-time, kills redundant converts).
// Kernel 1: QKV projection; 1 wave per 16 rows owns all 12 (QKV x 4 h-tiles) accumulators,
//           so each x A-fragment convert feeds 12 v_wmma_f32_16x16x32_bf16.
// Kernel 2: flash attention, one wave per 16 query rows, online softmax fp32,
//           P transposed C->A layout via per-wave LDS.
// Workspace: Q,K row-major bf16 + V^T bf16 (6 MB) + Wt bf16 (384 KB).

typedef __attribute__((ext_vector_type(16))) __bf16          v16bf;
typedef __attribute__((ext_vector_type(16))) unsigned short  v16us;
typedef __attribute__((ext_vector_type(8)))  float           v8f;

#define BATCH  4
#define SEQ    4096
#define DMODEL 1024
#define DHEAD  64

// Hardware bf16 convert (clang lowers fptrunc f32->bf16 natively on gfx1250).
__device__ __forceinline__ unsigned short f2bf(float f) {
  return __builtin_bit_cast(unsigned short, (__bf16)f);
}

__device__ __forceinline__ v8f wmma_bf16(v16us a, v16us b, v8f c) {
  return __builtin_amdgcn_wmma_f32_16x16x32_bf16(
      false, __builtin_bit_cast(v16bf, a),
      false, __builtin_bit_cast(v16bf, b),
      (short)0, c, false, false);
}

// Load one 16x32 bf16 A/B fragment half-row: elems 0..7 = base[koff..koff+7],
// elems 8..15 = base[koff+16..koff+23]  (per CDNA5 16-bit operand VGPR layout).
__device__ __forceinline__ v16us load_frag(const unsigned short* base, int koff) {
  union { uint4 v[2]; unsigned short us[16]; } t;
  t.v[0] = *(const uint4*)(base + koff);
  t.v[1] = *(const uint4*)(base + koff + 16);
  v16us r;
#pragma unroll
  for (int i = 0; i < 16; ++i) r[i] = t.us[i];
  return r;
}

// ---------------- Kernel 0: W -> bf16, transposed ----------------
__global__ __launch_bounds__(256, 1) void wprep_kernel(
    const float* __restrict__ Wq, const float* __restrict__ Wk,
    const float* __restrict__ Wv, unsigned short* __restrict__ Wt)
{
  const int idx = blockIdx.x * 256 + threadIdx.x;   // 0 .. 64*1024-1
  const int z = blockIdx.y;                         // 0=Q 1=K 2=V
  const int h = idx >> 10;
  const int k = idx & (DMODEL - 1);
  const float* W = (z == 0) ? Wq : (z == 1) ? Wk : Wv;
  Wt[(size_t)z * DHEAD * DMODEL + (size_t)h * DMODEL + k] = f2bf(W[k * DHEAD + h]);
}

// ---------------- Kernel 1: fused QKV projection ----------------
__global__ __launch_bounds__(32, 1) void qkv_kernel(
    const float* __restrict__ x,
    const unsigned short* __restrict__ Wt,
    const float* __restrict__ bq, const float* __restrict__ bk,
    const float* __restrict__ bv,
    unsigned short* __restrict__ Qb, unsigned short* __restrict__ Kb,
    unsigned short* __restrict__ Vt)
{
  const int lane = threadIdx.x & 31;
  const int col  = lane & 15;
  const int hi   = lane >> 4;
  const int koff = hi * 8;
  const int sTile = blockIdx.x * 16;
  const int b     = blockIdx.y;

  v8f acc[12];                       // [0..3]=Q h-tiles, [4..7]=K, [8..11]=V
#pragma unroll
  for (int ht = 0; ht < 4; ++ht) {
    const int h = ht * 16 + col;
    const float bqv = bq[h], bkv = bk[h], bvv = bv[h];
#pragma unroll
    for (int r = 0; r < 8; ++r) {
      acc[ht][r] = bqv; acc[4 + ht][r] = bkv; acc[8 + ht][r] = bvv;
    }
  }

  const float* xrow = x + ((size_t)b * SEQ + sTile + col) * DMODEL;

  for (int k0 = 0; k0 < DMODEL; k0 += 32) {
    // A fragment from x: converted ONCE, amortized over 12 WMMAs
    union { float4 v[4]; float f[16]; } ax;
    const float4* xp = (const float4*)(xrow + k0 + koff);
    const float4* xq = (const float4*)(xrow + k0 + koff + 16);
    ax.v[0] = xp[0]; ax.v[1] = xp[1]; ax.v[2] = xq[0]; ax.v[3] = xq[1];
    v16us a;
#pragma unroll
    for (int i = 0; i < 16; ++i) a[i] = f2bf(ax.f[i]);

#pragma unroll
    for (int z = 0; z < 3; ++z)
#pragma unroll
      for (int ht = 0; ht < 4; ++ht) {
        const unsigned short* wrow =
            Wt + ((size_t)z * DHEAD + ht * 16 + col) * DMODEL + k0;
        acc[z * 4 + ht] = wmma_bf16(a, load_frag(wrow, koff), acc[z * 4 + ht]);
      }
  }

  // C-layout: lane holds col h, rows (r + 8*hi)
  const int rowbase = sTile + hi * 8;
#pragma unroll
  for (int ht = 0; ht < 4; ++ht) {
    const int h = ht * 16 + col;
#pragma unroll
    for (int r = 0; r < 8; ++r) {
      const size_t idx = ((size_t)b * SEQ + rowbase + r) * DHEAD + h;
      Qb[idx] = f2bf(acc[ht][r]);
      Kb[idx] = f2bf(acc[4 + ht][r]);
    }
    union { unsigned short us[8]; uint4 v; } vp;
#pragma unroll
    for (int r = 0; r < 8; ++r) vp.us[r] = f2bf(acc[8 + ht][r]);
    *(uint4*)(Vt + ((size_t)b * DHEAD + h) * SEQ + rowbase) = vp.v;
  }
}

// ---------------- Kernel 2: flash attention ----------------
__global__ __launch_bounds__(256, 1) void attn_kernel(
    const unsigned short* __restrict__ Qb,
    const unsigned short* __restrict__ Kb,
    const unsigned short* __restrict__ Vt,
    float* __restrict__ out)
{
  __shared__ float lds[8][16 * 33];        // per-wave P transpose buffer
  const int lane = threadIdx.x & 31;
  const int wave = threadIdx.x >> 5;       // 0..7
  const int col  = lane & 15;
  const int hi   = lane >> 4;
  const int koff = hi * 8;
  const int b     = blockIdx.y;
  const int qbase = blockIdx.x * 128 + wave * 16;

  // Q A-fragments: lane = query row (qbase+col), h in [0,32) and [32,64)
  const unsigned short* qrow = Qb + ((size_t)b * SEQ + qbase + col) * DHEAD;
  const v16us a0 = load_frag(qrow, koff);
  const v16us a1 = load_frag(qrow + 32, koff);

  float m_i[8], l_i[8];
  v8f acc[4];
#pragma unroll
  for (int r = 0; r < 8; ++r) { m_i[r] = -__builtin_inff(); l_i[r] = 0.f; }
#pragma unroll
  for (int f = 0; f < 4; ++f)
#pragma unroll
    for (int r = 0; r < 8; ++r) acc[f][r] = 0.f;

  float* pt = lds[wave];
  const size_t kbbase = (size_t)b * SEQ * DHEAD;

  for (int j = 0; j < SEQ; j += 32) {
    // scores: B-fragments = K rows (lane = key), contracted over h
    const unsigned short* krow0 = Kb + kbbase + (size_t)(j + col) * DHEAD;
    const unsigned short* krow1 = krow0 + 16 * DHEAD;
    if (j + 32 < SEQ) {
      __builtin_prefetch(krow0 + 32 * DHEAD, 0, 1);
      __builtin_prefetch(krow1 + 32 * DHEAD, 0, 1);
    }
    v8f s0 = {}, s1 = {};
    s0 = wmma_bf16(a0, load_frag(krow0, koff),      s0);
    s0 = wmma_bf16(a1, load_frag(krow0 + 32, koff), s0);
    s1 = wmma_bf16(a0, load_frag(krow1, koff),      s1);
    s1 = wmma_bf16(a1, load_frag(krow1 + 32, koff), s1);

    // online softmax (C-layout: lane holds rows r+8*hi, one key column each)
    float mt[8];
#pragma unroll
    for (int r = 0; r < 8; ++r) {
      s0[r] *= 0.125f; s1[r] *= 0.125f;     // 1/sqrt(64)
      mt[r] = fmaxf(s0[r], s1[r]);
    }
#pragma unroll
    for (int mask = 1; mask < 16; mask <<= 1)
#pragma unroll
      for (int r = 0; r < 8; ++r)
        mt[r] = fmaxf(mt[r], __shfl_xor(mt[r], mask, 32));

    float corr[8], rs[8];
#pragma unroll
    for (int r = 0; r < 8; ++r) {
      const float mn = fmaxf(m_i[r], mt[r]);
      corr[r] = __expf(m_i[r] - mn);
      m_i[r]  = mn;
      s0[r] = __expf(s0[r] - mn);
      s1[r] = __expf(s1[r] - mn);
      rs[r] = s0[r] + s1[r];
    }
#pragma unroll
    for (int mask = 1; mask < 16; mask <<= 1)
#pragma unroll
      for (int r = 0; r < 8; ++r)
        rs[r] += __shfl_xor(rs[r], mask, 32);
#pragma unroll
    for (int r = 0; r < 8; ++r) l_i[r] = l_i[r] * corr[r] + rs[r];
#pragma unroll
    for (int f = 0; f < 4; ++f)
#pragma unroll
      for (int r = 0; r < 8; ++r) acc[f][r] *= corr[r];

    // transpose P (C-layout) -> A-layout through per-wave LDS
    const int rowb = hi * 8;
#pragma unroll
    for (int r = 0; r < 8; ++r) {
      pt[(rowb + r) * 33 + col]      = s0[r];
      pt[(rowb + r) * 33 + 16 + col] = s1[r];
    }
    asm volatile("s_wait_dscnt 0" ::: "memory");   // LDS in-order per wave; fence re-read
    v16us pa;
    const float* prow = pt + col * 33;
#pragma unroll
    for (int i = 0; i < 8; ++i) {
      pa[i]     = f2bf(prow[koff + i]);
      pa[8 + i] = f2bf(prow[koff + 16 + i]);
    }

    // O += P * V : B-fragments from V^T (lane = head col, contiguous keys)
#pragma unroll
    for (int f = 0; f < 4; ++f) {
      const unsigned short* vrow =
          Vt + ((size_t)b * DHEAD + f * 16 + col) * SEQ + j;
      acc[f] = wmma_bf16(pa, load_frag(vrow, koff), acc[f]);
    }
  }

  // normalize and store fp32 [B,S,64]
#pragma unroll
  for (int r = 0; r < 8; ++r) {
    const float inv = 1.0f / l_i[r];
    const int row = qbase + hi * 8 + r;
#pragma unroll
    for (int f = 0; f < 4; ++f)
      out[((size_t)b * SEQ + row) * DHEAD + f * 16 + col] = acc[f][r] * inv;
  }
}

extern "C" void kernel_launch(void* const* d_in, const int* in_sizes, int n_in,
                              void* d_out, int out_size, void* d_ws, size_t ws_size,
                              hipStream_t stream) {
  (void)in_sizes; (void)n_in; (void)out_size; (void)ws_size;
  const float* x  = (const float*)d_in[0];
  const float* Wq = (const float*)d_in[1];
  const float* bq = (const float*)d_in[2];
  const float* Wk = (const float*)d_in[3];
  const float* bk = (const float*)d_in[4];
  const float* Wv = (const float*)d_in[5];
  const float* bv = (const float*)d_in[6];

  // workspace: Q, K row-major bf16 + V^T bf16 (6 MB) + Wt bf16 (384 KB)
  unsigned short* Qb = (unsigned short*)d_ws;
  unsigned short* Kb = Qb + (size_t)BATCH * SEQ * DHEAD;
  unsigned short* Vt = Kb + (size_t)BATCH * SEQ * DHEAD;
  unsigned short* Wt = Vt + (size_t)BATCH * SEQ * DHEAD;

  dim3 g0(DHEAD * DMODEL / 256, 3);
  wprep_kernel<<<g0, dim3(256), 0, stream>>>(Wq, Wk, Wv, Wt);

  dim3 g1(SEQ / 16, BATCH);
  qkv_kernel<<<g1, dim3(32), 0, stream>>>(x, Wt, bq, bk, bv, Qb, Kb, Vt);

  dim3 g2(SEQ / 128, BATCH);
  attn_kernel<<<g2, dim3(256), 0, stream>>>(Qb, Kb, Vt, (float*)d_out);
}
